// Gemma3MoEAttention_26800595927638
// MI455X (gfx1250) — matrix operand order repaired and checked by reading, not compiled, not run on hardware
//
#include <hip/hip_runtime.h>
#include <stdint.h>

// ---------------------------------------------------------------------------
// Types for CDNA5 WMMA (gfx1250, wave32)
// ---------------------------------------------------------------------------
typedef __attribute__((ext_vector_type(16))) __bf16 bf16x16;
typedef __attribute__((ext_vector_type(8)))  __bf16 bf16x8;
typedef __attribute__((ext_vector_type(8)))  float  f32x8;

union BF16x16U { bf16x16 v; bf16x8 h[2]; };

__device__ inline f32x8 wmma_bf16(bf16x16 a, bf16x16 b, f32x8 c) {
  // D(f32,16x16) = A(bf16,16x32) * B(bf16,32x16) + C
  return __builtin_amdgcn_wmma_f32_16x16x32_bf16(
      /*neg_a=*/false, a, /*neg_b=*/false, b,
      /*c_mod=*/(short)0, c, /*reuse_a=*/false, /*reuse_b=*/false);
}

__device__ inline bf16x16 ld_frag2(const __bf16* lo, const __bf16* hi) {
  BF16x16U u;
  u.h[0] = *(const bf16x8*)lo;
  u.h[1] = *(const bf16x8*)hi;
  return u.v;
}

// Async global -> LDS copy of 16 bytes per lane (ASYNCcnt-tracked).
// Flat LDS addr[31:0] is the LDS byte offset, so truncating the generic
// pointer gives the DS address for VDST.
__device__ inline void async_ld_b128(const void* gp, void* lp) {
  unsigned la = (unsigned)(uintptr_t)lp;
  uint64_t ga = (uint64_t)(uintptr_t)gp;
  asm volatile("global_load_async_to_lds_b128 %0, %1, off"
               :: "v"(la), "v"(ga) : "memory");
}
__device__ inline void async_wait0() {
  asm volatile("s_wait_asynccnt 0x0" ::: "memory");
}

// Problem constants
#define TT      4096
#define DD      2560
#define NH      8
#define KH      4
#define HH      256
#define WINDOW  1024
#define QKV_C   4096        // N*H + 2*K*H output columns
#define NEG_BIG (-3.0e38f)

// ---------------------------------------------------------------------------
// Prep kernels: bf16 casts + weight transposes
// ---------------------------------------------------------------------------
__global__ __launch_bounds__(256) void cvt_x_kernel(
    const float* __restrict__ x1, const float* __restrict__ x2,
    __bf16* __restrict__ xb) {
  size_t i = (size_t)blockIdx.x * 256 + threadIdx.x;
  if (i >= (size_t)TT * DD) return;
  size_t t = i / DD;
  float v = (t < 2048) ? x1[i] : x2[i - (size_t)2048 * DD];
  xb[i] = (__bf16)v;
}

__global__ __launch_bounds__(256) void prep_wt_kernel(
    const float* __restrict__ qw, const float* __restrict__ kvw,
    __bf16* __restrict__ wt) {
  // wt[col][d] (K-contiguous for B-fragments), col in [0, 4096)
  size_t i = (size_t)blockIdx.x * 256 + threadIdx.x;
  if (i >= (size_t)QKV_C * DD) return;
  int col = (int)(i / DD), d = (int)(i % DD);
  float v;
  if (col < NH * HH) {
    int n = col >> 8, h = col & 255;
    v = qw[((size_t)n * DD + d) * HH + h];
  } else {
    int c = col - NH * HH;        // [0, 2048): k then v
    int grp = c >> 10;            // 0 = k, 1 = v
    int kh  = (c >> 8) & 3;
    int h   = c & 255;
    v = kvw[(((size_t)grp * KH + kh) * DD + d) * HH + h];
  }
  wt[i] = (__bf16)v;
}

__global__ __launch_bounds__(256) void prep_owt_kernel(
    const float* __restrict__ ow, __bf16* __restrict__ owt) {
  // owt[d][c] where c = n*H + h; ow[n][h][d] = ow[c*D + d]
  size_t i = (size_t)blockIdx.x * 256 + threadIdx.x;
  if (i >= (size_t)DD * (NH * HH)) return;
  int d = (int)(i / (NH * HH)), c = (int)(i % (NH * HH));
  owt[i] = (__bf16)ow[(size_t)c * DD + d];
}

// ---------------------------------------------------------------------------
// Tiled bf16 GEMM: C(MxN,f32) = A(MxK,bf16 row-major) * Bt(NxK,bf16)^T
// BM=128, BN=128, BK=32; 8 waves; each wave 32x64 via 2x4 WMMA.
// Double-buffered LDS; tiles staged with GLOBAL_LOAD_ASYNC_TO_LDS_B128
// overlapped with WMMA compute; one barrier + one ASYNCcnt wait per K-step.
// ---------------------------------------------------------------------------
#define G_BM 128
#define G_BN 128
#define G_BK 32
#define G_AS 40   // LDS row stride (elements), padded
#define G_BS 40

template <int MM, int NC, int KD>
__global__ __launch_bounds__(256) void gemm_bf16_kernel(
    const __bf16* __restrict__ A, const __bf16* __restrict__ Bt,
    float* __restrict__ C) {
  __shared__ __align__(16) __bf16 As[2][G_BM * G_AS];
  __shared__ __align__(16) __bf16 Bs[2][G_BN * G_BS];

  const int tid  = threadIdx.x;
  const int wave = tid >> 5;
  const int lane = tid & 31;
  const int lrow = lane & 15;
  const int lhi  = lane >> 4;
  const int m0 = blockIdx.x * G_BM;
  const int n0 = blockIdx.y * G_BN;
  const int wm = (wave >> 1) * 32;   // 4 wave-groups along M (32 rows each)
  const int wn = (wave & 1) * 64;    // 2 wave-groups along N (64 cols each)

  // Static staging coordinates: each thread moves 2x16B of A and 2x16B of B
  const int ar = tid >> 1, ako = (tid & 1) * 16;   // A: 128 rows x 32 k
  const int bc = tid >> 1, bko = (tid & 1) * 16;   // B: 128 cols x 32 k

  auto stage = [&](int k0, int b) {
    const __bf16* gp = A + (size_t)(m0 + ar) * KD + k0 + ako;
    async_ld_b128(gp,     &As[b][ar * G_AS + ako]);
    async_ld_b128(gp + 8, &As[b][ar * G_AS + ako + 8]);
    const __bf16* hp = Bt + (size_t)(n0 + bc) * KD + k0 + bko;
    async_ld_b128(hp,     &Bs[b][bc * G_BS + bko]);
    async_ld_b128(hp + 8, &Bs[b][bc * G_BS + bko + 8]);
  };

  f32x8 acc[2][4];
#pragma unroll
  for (int a = 0; a < 2; a++)
#pragma unroll
    for (int b = 0; b < 4; b++)
#pragma unroll
      for (int j = 0; j < 8; j++) acc[a][b][j] = 0.f;

  stage(0, 0);                      // prologue: tile 0 -> buffer 0
  int buf = 0;
  for (int k0 = 0; k0 < KD; k0 += G_BK) {
    async_wait0();                  // tile k0 landed (this wave's chunks)
    __syncthreads();                // ...landed everywhere; buf^1 free
    if (k0 + G_BK < KD) stage(k0 + G_BK, buf ^ 1);  // overlaps compute

    bf16x16 af[2], bfm[4];
#pragma unroll
    for (int mi = 0; mi < 2; mi++) {
      const __bf16* p = &As[buf][(wm + mi * 16 + lrow) * G_AS];
      af[mi] = ld_frag2(p + lhi * 8, p + 16 + lhi * 8);       // A layout
    }
#pragma unroll
    for (int ni = 0; ni < 4; ni++) {
      const __bf16* p = &Bs[buf][(wn + ni * 16 + lrow) * G_BS];
      bfm[ni] = ld_frag2(p + lhi * 8, p + 16 + lhi * 8);       // B layout
    }
#pragma unroll
    for (int mi = 0; mi < 2; mi++)
#pragma unroll
      for (int ni = 0; ni < 4; ni++)
        acc[mi][ni] = wmma_bf16(af[mi], bfm[ni], acc[mi][ni]);
    buf ^= 1;
  }

  // epilogue: C element (M,N): M = base + 8*lhi + j, N = base + lrow
#pragma unroll
  for (int mi = 0; mi < 2; mi++)
#pragma unroll
    for (int ni = 0; ni < 4; ni++) {
      const int row = m0 + wm + mi * 16 + lhi * 8;
      const int col = n0 + wn + ni * 16 + lrow;
      float* cp = C + (size_t)row * NC + col;
#pragma unroll
      for (int j = 0; j < 8; j++) cp[j * NC] = acc[mi][ni][j];
    }
}

// ---------------------------------------------------------------------------
// RMS-norm + RoPE + q-scale; emits bf16 q/k and transposed bf16 V
// grid: (T, 16): slots 0..7 = q heads, 8..11 = k heads, 12..15 = v heads
// ---------------------------------------------------------------------------
__global__ __launch_bounds__(256) void normrope_kernel(
    const float* __restrict__ qkv, const float* __restrict__ qns,
    const float* __restrict__ kns, const int* __restrict__ positions,
    __bf16* __restrict__ qb, __bf16* __restrict__ kb,
    __bf16* __restrict__ vbt) {
  const int t = blockIdx.x;
  const int slot = blockIdx.y;
  const int h = threadIdx.x;
  __shared__ float red[8];
  __shared__ float xs[HH];

  int colbase;
  if (slot < 8)       colbase = slot * HH;
  else if (slot < 12) colbase = NH * HH + (slot - 8) * HH;
  else                colbase = NH * HH + KH * HH + (slot - 12) * HH;

  float v = qkv[(size_t)t * QKV_C + colbase + h];

  if (slot >= 12) {  // V: no norm/rope, store transposed [kh][h][t]
    int kh = slot - 12;
    vbt[((size_t)(kh * HH + h)) * TT + t] = (__bf16)v;
    return;
  }

  // RMS norm over H = 256
  float ss = v * v;
#pragma unroll
  for (int off = 16; off > 0; off >>= 1) ss += __shfl_xor(ss, off, 32);
  if ((h & 31) == 0) red[h >> 5] = ss;
  __syncthreads();
  float tot = 0.f;
#pragma unroll
  for (int i = 0; i < 8; i++) tot += red[i];
  float rn = rsqrtf(tot * (1.0f / (float)HH) + 1e-6f);
  float sc = (slot < 8) ? qns[h] : kns[h];
  float x = v * rn * (1.0f + sc);
  xs[h] = x;
  __syncthreads();

  // RoPE
  int pos = positions[t];
  int i = (h < 128) ? h : (h - 128);
  float frac = (float)i * (1.0f / 128.0f);
  float ts = __powf(10000.0f, frac);
  float ang = (float)pos / ts;
  float sn, cs;
  __sincosf(ang, &sn, &cs);
  float other = (h < 128) ? xs[h + 128] : xs[h - 128];
  float out = (h < 128) ? (x * cs - other * sn) : (x * cs + other * sn);

  if (slot < 8) {
    out *= 0.0625f;  // H^-0.5 = 1/16
    qb[(size_t)t * (NH * HH) + slot * HH + h] = (__bf16)out;
  } else {
    kb[(size_t)t * (KH * HH) + (slot - 8) * HH + h] = (__bf16)out;
  }
}

// ---------------------------------------------------------------------------
// Flash attention with sliding window, bf16 WMMA
// grid: (T/128, N); block = 256 threads (8 waves); wave w owns rows w*16..+15
// K/V tiles staged via GLOBAL_LOAD_ASYNC_TO_LDS_B128.
// ---------------------------------------------------------------------------
#define A_BM 128
#define A_BS 32
#define K_STR 264   // 32 keys x 256 h, padded
#define V_STR 40    // 256 h x 32 keys, padded
#define P_STR 40    // 16 rows x 32 keys, padded

__global__ __launch_bounds__(256) void attn_kernel(
    const __bf16* __restrict__ qb, const __bf16* __restrict__ kb,
    const __bf16* __restrict__ vbt, __bf16* __restrict__ encb) {
  __shared__ __align__(16) __bf16 Ks[A_BS * K_STR];
  __shared__ __align__(16) __bf16 Vs[HH * V_STR];
  __shared__ __align__(16) __bf16 Ps[8 * 16 * P_STR];

  const int tid  = threadIdx.x;
  const int wave = tid >> 5;
  const int lane = tid & 31;
  const int lrow = lane & 15;
  const int lhi  = lane >> 4;
  const int n  = blockIdx.y;
  const int kh = n >> 1;               // G = N/K = 2
  const int t0 = blockIdx.x * A_BM;

  // Q fragments for this wave's 16 rows (A-frag row = lane%16)
  const int qrow = t0 + wave * 16 + lrow;
  bf16x16 qf[8];
  {
    const __bf16* qp = qb + (size_t)qrow * (NH * HH) + n * HH;
#pragma unroll
    for (int hk = 0; hk < 8; hk++)
      qf[hk] = ld_frag2(qp + hk * 32 + lhi * 8, qp + hk * 32 + 16 + lhi * 8);
  }

  f32x8 o[16];
#pragma unroll
  for (int ht = 0; ht < 16; ht++)
#pragma unroll
    for (int j = 0; j < 8; j++) o[ht][j] = 0.f;
  float rmax[8], rsum[8];
#pragma unroll
  for (int j = 0; j < 8; j++) { rmax[j] = NEG_BIG; rsum[j] = 0.f; }

  int sl = t0 - WINDOW + 1;
  if (sl < 0) sl = 0;
  sl &= ~31;

  for (int s0 = sl; s0 <= t0 + A_BM - 1; s0 += A_BS) {
    // async stage K tile (32 keys x 256 h, natural) and V^T tile (256 x 32)
#pragma unroll
    for (int c = 0; c < 4; c++) {
      int cc = tid + c * 256;
      int r = cc >> 5, ko = (cc & 31) * 8;
      async_ld_b128(kb + (size_t)(s0 + r) * (KH * HH) + kh * HH + ko,
                    &Ks[r * K_STR + ko]);
    }
#pragma unroll
    for (int c = 0; c < 4; c++) {
      int cc = tid + c * 256;
      int hh = cc >> 2, so = (cc & 3) * 8;
      async_ld_b128(vbt + ((size_t)(kh * HH + hh)) * TT + s0 + so,
                    &Vs[hh * V_STR + so]);
    }
    async_wait0();
    __syncthreads();

    // S = Q K^T : 16x32 per wave (2 column fragments)
    f32x8 sfr[2];
#pragma unroll
    for (int ni = 0; ni < 2; ni++) {
#pragma unroll
      for (int j = 0; j < 8; j++) sfr[ni][j] = 0.f;
      const __bf16* bp = &Ks[(ni * 16 + lrow) * K_STR];
#pragma unroll
      for (int hk = 0; hk < 8; hk++) {
        bf16x16 kf = ld_frag2(bp + hk * 32 + lhi * 8,
                              bp + hk * 32 + 16 + lhi * 8);
        sfr[ni] = wmma_bf16(qf[hk], kf, sfr[ni]);
      }
    }

    // mask + online softmax (row r lives in 16 lanes of one half-wave)
    float pkeep[2][8];
#pragma unroll
    for (int j = 0; j < 8; j++) {
      int r = t0 + wave * 16 + lhi * 8 + j;
      int sA = s0 + lrow, sB = s0 + 16 + lrow;
      float v0 = ((sA <= r) && (sA + WINDOW > r)) ? sfr[0][j] : NEG_BIG;
      float v1 = ((sB <= r) && (sB + WINDOW > r)) ? sfr[1][j] : NEG_BIG;
      float m = fmaxf(v0, v1);
#pragma unroll
      for (int off = 8; off > 0; off >>= 1)
        m = fmaxf(m, __shfl_xor(m, off, 32));
      float mx = fmaxf(rmax[j], m);
      float scale = __expf(rmax[j] - mx);
      float p0 = (v0 <= -1e37f) ? 0.f : __expf(v0 - mx);
      float p1 = (v1 <= -1e37f) ? 0.f : __expf(v1 - mx);
      float rs = p0 + p1;
#pragma unroll
      for (int off = 8; off > 0; off >>= 1) rs += __shfl_xor(rs, off, 32);
      rsum[j] = rsum[j] * scale + rs;
      rmax[j] = mx;
      pkeep[0][j] = p0;
      pkeep[1][j] = p1;
#pragma unroll
      for (int ht = 0; ht < 16; ht++) o[ht][j] *= scale;
    }

    // P (C-layout) -> wave-private LDS -> A-fragment
    {
      __bf16* pp = &Ps[wave * 16 * P_STR];
#pragma unroll
      for (int ni = 0; ni < 2; ni++)
#pragma unroll
        for (int j = 0; j < 8; j++)
          pp[(lhi * 8 + j) * P_STR + ni * 16 + lrow] = (__bf16)pkeep[ni][j];
    }
    bf16x16 pf;
    {
      const __bf16* p = &Ps[(wave * 16 + lrow) * P_STR];
      pf = ld_frag2(p + lhi * 8, p + 16 + lhi * 8);
    }

    // O += P @ V  (16 h-tiles of 16 columns)
#pragma unroll
    for (int ht = 0; ht < 16; ht++) {
      const __bf16* vp = &Vs[(ht * 16 + lrow) * V_STR];
      bf16x16 vf = ld_frag2(vp + lhi * 8, vp + 16 + lhi * 8);
      o[ht] = wmma_bf16(pf, vf, o[ht]);
    }
    __syncthreads();
  }

  // normalize, store enc (bf16) — base pointer + constant offsets
  {
    const int r0 = t0 + wave * 16 + lhi * 8;
    __bf16* bp = encb + (size_t)r0 * (NH * HH) + n * HH + lrow;
#pragma unroll
    for (int j = 0; j < 8; j++) {
      float invl = 1.0f / rsum[j];
#pragma unroll
      for (int ht = 0; ht < 16; ht++)
        bp[j * (NH * HH) + ht * 16] = (__bf16)(o[ht][j] * invl);
    }
  }
}

// ---------------------------------------------------------------------------
// Launch
// ---------------------------------------------------------------------------
extern "C" void kernel_launch(void* const* d_in, const int* in_sizes, int n_in,
                              void* d_out, int out_size, void* d_ws,
                              size_t ws_size, hipStream_t stream) {
  const float* x1  = (const float*)d_in[0];
  const float* x2  = (const float*)d_in[1];
  const float* qw  = (const float*)d_in[2];
  const float* kvw = (const float*)d_in[3];
  const float* ow  = (const float*)d_in[4];
  const float* qns = (const float*)d_in[5];
  const float* kns = (const float*)d_in[6];
  const int* positions = (const int*)d_in[7];
  // d_in[8] = attn_mask (causal tril) — computed analytically in-kernel.

  char* ws = (char*)d_ws;
  size_t off = 0;
  float*  qkv = (float*)(ws + off);  off += (size_t)TT * QKV_C * 4;       // 67 MB
  __bf16* xb  = (__bf16*)(ws + off); off += (size_t)TT * DD * 2;          // 21 MB
  __bf16* wt  = (__bf16*)(ws + off); off += (size_t)QKV_C * DD * 2;       // 21 MB
  __bf16* owt = (__bf16*)(ws + off); off += (size_t)DD * NH * HH * 2;     // 10.5 MB
  __bf16* qb  = (__bf16*)(ws + off); off += (size_t)TT * NH * HH * 2;     // 16.8 MB
  __bf16* kb  = (__bf16*)(ws + off); off += (size_t)TT * KH * HH * 2;     // 8.4 MB
  __bf16* vbt = (__bf16*)(ws + off); off += (size_t)TT * KH * HH * 2;     // 8.4 MB
  __bf16* encb = (__bf16*)(ws + off);                                      // 16.8 MB
  float* out = (float*)d_out;

  // 1) casts + weight transposes
  cvt_x_kernel<<<(TT * DD + 255) / 256, 256, 0, stream>>>(x1, x2, xb);
  prep_wt_kernel<<<(QKV_C * DD + 255) / 256, 256, 0, stream>>>(qw, kvw, wt);
  prep_owt_kernel<<<(DD * NH * HH + 255) / 256, 256, 0, stream>>>(ow, owt);

  // 2) fused QKV projection: qkv[T x 4096] = xb[T x D] * wt^T
  dim3 g1(TT / G_BM, QKV_C / G_BN);
  gemm_bf16_kernel<TT, QKV_C, DD><<<g1, 256, 0, stream>>>(xb, wt, qkv);

  // 3) RMS-norm + RoPE + scale -> bf16 q/k, transposed bf16 V
  dim3 g2(TT, 16);
  normrope_kernel<<<g2, 256, 0, stream>>>(qkv, qns, kns, positions, qb, kb, vbt);

  // 4) sliding-window flash attention -> enc bf16
  dim3 g3(TT / A_BM, NH);
  attn_kernel<<<g3, 256, 0, stream>>>(qb, kb, vbt, encb);

  // 5) output projection: out[T x D] = encb[T x 2048] * owt^T
  dim3 g4(TT / G_BM, DD / G_BN);
  gemm_bf16_kernel<TT, DD, NH * HH><<<g4, 256, 0, stream>>>(encb, owt, out);
}
